// ColorToneMapper_13829794693364
// MI455X (gfx1250) — matrix operand order repaired and checked by reading, not compiled, not run on hardware
//
#include <hip/hip_runtime.h>
#include <hip/hip_bf16.h>
#include <stdint.h>

// ---------------------------------------------------------------------------
// MI455X (gfx1250) tiny-MLP tone mapper.
// Compute-bound (138 GFLOP vs 16MB HBM traffic) -> drive v_wmma_f32_16x16x32_f16.
// ---------------------------------------------------------------------------

typedef __attribute__((ext_vector_type(16))) _Float16 v16h;
typedef __attribute__((ext_vector_type(8)))  float    v8f;

union ABFrag {
    v16h  h;
    uint4 u[2];
};

__device__ __forceinline__ uint32_t packh(_Float16 lo, _Float16 hi) {
    unsigned short a = __builtin_bit_cast(unsigned short, lo);
    unsigned short b = __builtin_bit_cast(unsigned short, hi);
    return (uint32_t)a | ((uint32_t)b << 16);
}

// ---------------------------------------------------------------------------
// Prep: convert W1/W2 (128x128 f32, row-major [k][n]) into f16 packed in the
// contiguous-k WMMA B-fragment order (matches the A loads) so the main kernel streams fragments with
// two coalesced b128 loads per lane.
//
// B fragment (32x16, f16, 8 VGPRs): lane l -> column n = l&15, koff=(l>>4)*16,
// VGPR j holds K = koff+2j (lo 16b) and koff+2j+1 (hi 16b).
// Fragment for (kb, nb) stored at uint index ((kb*8+nb)*32 + lane)*8 + j.
// 32 tiles * 32 lanes * 8 uints = 8192 uints = 32 KB per layer.
// ---------------------------------------------------------------------------
__global__ __launch_bounds__(256) void prep_weights_kernel(
        const float* __restrict__ W1, const float* __restrict__ W2,
        uint32_t* __restrict__ W1f, uint32_t* __restrict__ W2f) {
    const int idx  = blockIdx.x * blockDim.x + threadIdx.x;   // 0..8191
    const int j    = idx & 7;
    const int lane = (idx >> 3) & 31;
    const int t    = idx >> 8;            // tile 0..31
    const int kb   = t >> 3;
    const int nb   = t & 7;
    const int n    = lane & 15;
    const int koff = (lane >> 4) * 16;
    const int kk   = kb * 32 + koff + 2 * j;
    const int col  = nb * 16 + n;

    W1f[idx] = packh((_Float16)W1[kk * 128 + col], (_Float16)W1[(kk + 1) * 128 + col]);
    W2f[idx] = packh((_Float16)W2[kk * 128 + col], (_Float16)W2[(kk + 1) * 128 + col]);
}

// ---------------------------------------------------------------------------
// Main kernel: 128 threads = 4 waves; each wave owns 16 rays.
// Activations staged in LDS as f16, row stride 136 halves (272B = 68 dwords,
// 68 mod 64 = 4 -> consecutive rows rotate 4 banks: conflict-free tiles).
// ---------------------------------------------------------------------------
#define ROWSTRIDE 136
#define WAVES 4

__global__ __launch_bounds__(128) void mlp_wmma_kernel(
        const float* __restrict__ rad,
        const float* __restrict__ W0, const float* __restrict__ b0,
        const uint32_t* __restrict__ W1f, const float* __restrict__ b1,
        const uint32_t* __restrict__ W2f, const float* __restrict__ b2,
        const float* __restrict__ W3, const float* __restrict__ b3,
        float* __restrict__ out) {
    __shared__ __align__(16) _Float16 acts[WAVES][2][16 * ROWSTRIDE];

    const int lane    = threadIdx.x & 31;
    const int wave    = threadIdx.x >> 5;
    const int rayBase = (blockIdx.x * WAVES + wave) * 16;
    const int row     = lane & 15;     // M for A-frags, N for C/D frags
    const int half    = lane >> 4;     // which half of the wave

    // ---- Layer 0: h0 = relu(log(max(r,eps)) * W0 + b0), f16 into LDS ------
    {
        const float r  = rad[rayBase + row];
        const float lr = __logf(fmaxf(r, 1e-8f));
        const int   c0 = half * 64;
        _Float16* dst0 = &acts[wave][0][row * ROWSTRIDE];
        #pragma unroll 8
        for (int c = 0; c < 64; ++c) {
            float h = fmaxf(lr * W0[c0 + c] + b0[c0 + c], 0.0f);
            dst0[c0 + c] = (_Float16)h;
        }
    }
    asm volatile("s_wait_dscnt 0" ::: "memory");   // wave-private LDS handoff

    // ---- Layers 1 & 2: 128x128 via v_wmma_f32_16x16x32_f16 ----------------
    #pragma unroll
    for (int layer = 0; layer < 2; ++layer) {
        const uint32_t* __restrict__ Wf   = layer ? W2f : W1f;
        const float*    __restrict__ bias = layer ? b2  : b1;
        const _Float16* src = acts[wave][layer];
        _Float16*       dst = acts[wave][layer ^ 1];

        #pragma unroll
        for (int nb = 0; nb < 8; ++nb) {
            const float bv = bias[nb * 16 + row];
            v8f c = {bv, bv, bv, bv, bv, bv, bv, bv};

            #pragma unroll
            for (int kb = 0; kb < 4; ++kb) {
                // A fragment (16x32 f16): lane m=row holds K in
                // [kb*32+half*16, +16): two b128 LDS loads.
                ABFrag a;
                const uint4* ap = (const uint4*)(src + row * ROWSTRIDE + kb * 32 + half * 16);
                a.u[0] = ap[0];
                a.u[1] = ap[1];   // +8 halves = +1 uint4

                // B fragment: pre-swizzled, 32B contiguous per lane.
                ABFrag b;
                const uint4* bp = (const uint4*)(Wf + (((kb * 8 + nb) * 32) + lane) * 8);
                b.u[0] = bp[0];
                b.u[1] = bp[1];

                c = __builtin_amdgcn_wmma_f32_16x16x32_f16(
                        false, a.h, false, b.h, (short)0, c, false, false);
            }

            // D layout: VGPR r -> row m = half*8 + r, col n = nb*16 + (lane&15)
            #pragma unroll
            for (int r = 0; r < 8; ++r) {
                float v = fmaxf(c[r], 0.0f);
                dst[(half * 8 + r) * ROWSTRIDE + nb * 16 + row] = (_Float16)v;
            }
        }
        asm volatile("s_wait_dscnt 0" ::: "memory");
    }

    // ---- Layer 3: color = tanh(h @ W3 + b3) -------------------------------
    {
        const _Float16* src = acts[wave][0];
        float sum = 0.0f;
        const int c0 = half * 64;
        #pragma unroll 8
        for (int c = 0; c < 64; ++c) {
            sum += (float)src[row * ROWSTRIDE + c0 + c] * W3[c0 + c];
        }
        sum += __shfl_xor(sum, 16, 32);       // combine the two half-lane partials
        if (half == 0) {
            out[rayBase + row] = tanhf(sum + b3[0]);
        }
    }
}

// ---------------------------------------------------------------------------
extern "C" void kernel_launch(void* const* d_in, const int* in_sizes, int n_in,
                              void* d_out, int out_size, void* d_ws, size_t ws_size,
                              hipStream_t stream) {
    const float* rad = (const float*)d_in[0];
    // d_in[1] = x, d_in[2] = input_exps : unused by the reference
    const float* W0  = (const float*)d_in[3];
    const float* b0  = (const float*)d_in[4];
    const float* W1  = (const float*)d_in[5];
    const float* b1  = (const float*)d_in[6];
    const float* W2  = (const float*)d_in[7];
    const float* b2  = (const float*)d_in[8];
    const float* W3  = (const float*)d_in[9];
    const float* b3  = (const float*)d_in[10];

    uint32_t* W1f = (uint32_t*)d_ws;                       // 32 KB
    uint32_t* W2f = (uint32_t*)((char*)d_ws + 32768);      // 32 KB

    // One-shot weight conversion into WMMA B-fragment layout (8192 uints/layer).
    prep_weights_kernel<<<32, 256, 0, stream>>>(W1, W2, W1f, W2f);

    const int N = in_sizes[0];                 // 2,097,152 rays
    const int blocks = N / (16 * WAVES);       // 64 rays per block -> 32768 blocks
    mlp_wmma_kernel<<<blocks, 128, 0, stream>>>(
        rad, W0, b0, W1f, b1, W2f, b2, W3, b3, (float*)d_out);
}
